// VectorQuantizedVAE_77635828843071
// MI455X (gfx1250) — matrix operand order repaired and checked by reading, not compiled, not run on hardware
//
#include <hip/hip_runtime.h>

// ---------------------------------------------------------------------------
// CDNA5 (gfx1250) wave32 WMMA plumbing
// ---------------------------------------------------------------------------
typedef __attribute__((ext_vector_type(16))) _Float16 v16h;
typedef __attribute__((ext_vector_type(8)))  _Float16 v8h;   // 16B staging unit
typedef __attribute__((ext_vector_type(8)))  float    v8f;

__device__ __forceinline__ v8f wmma16(v16h a, v16h b, v8f c) {
  // D = A(16x32) * B(32x16) + C, fp32 accumulate
  return __builtin_amdgcn_wmma_f32_16x16x32_f16(false, a, false, b, (short)0, c,
                                                false, false);
}

union FragU { v16h v; unsigned u[8]; };

// A fragment: 16x32 f16, source row-major (row stride = ld halves).
// lane[0:16) row=lane,   K = 0..15
// lane[16:32) row=lane&15, K = 16..31
// -> u[0..7] contiguous 32B (matches frag_b)
__device__ __forceinline__ v16h frag_a(const _Float16* p, int ld) {
  const int lane = threadIdx.x & 31;
  const int row  = lane & 15;
  const int hf   = lane >> 4;
  FragU r;
  const unsigned* q = (const unsigned*)(p + (long)row * ld);
#pragma unroll
  for (int v = 0; v < 8; ++v) {
    const int kb = v * 2 + hf * 16;                        // even
    r.u[v] = q[kb >> 1];
  }
  return r.v;
}

// B fragment: 32x16 of B[k][n], sourced as W[n][k] row-major (stride ld).
// Contiguous k order, same as frag_a: lanes[0:16) hold K=0..15 (VGPR v -> K=2v,2v+1),
// lanes[16:32) hold K=16..31; N = lane&15.  u[0..7] contiguous 32B.
__device__ __forceinline__ v16h frag_b(const _Float16* p, int ld) {
  const int lane = threadIdx.x & 31;
  const int col  = lane & 15;
  const int hf   = lane >> 4;
  FragU r;
  const unsigned* q = (const unsigned*)(p + (long)col * ld);
#pragma unroll
  for (int v = 0; v < 8; ++v) {
    const int kb = v * 2 + hf * 16;                        // even
    r.u[v] = q[kb >> 1];
  }
  return r.v;
}
// C/D layout: lane holds col n = lane&15; VGPR j holds row m = j + 8*(lane>>4).

__device__ __forceinline__ v8h zero8h() {
  v8h z = {(_Float16)0.f, (_Float16)0.f, (_Float16)0.f, (_Float16)0.f,
           (_Float16)0.f, (_Float16)0.f, (_Float16)0.f, (_Float16)0.f};
  return z;
}

// ---------------------------------------------------------------------------
// Elementwise / conversion kernels
// ---------------------------------------------------------------------------
__global__ void cvt_f32_f16_kernel(const float* __restrict__ x,
                                   _Float16* __restrict__ y, long n) {
  for (long i = blockIdx.x * (long)blockDim.x + threadIdx.x; i < n;
       i += (long)gridDim.x * blockDim.x)
    y[i] = (_Float16)x[i];
}

__global__ void relu_f32_to_f16_kernel(const float* __restrict__ x,
                                       _Float16* __restrict__ y, long n) {
  for (long i = blockIdx.x * (long)blockDim.x + threadIdx.x; i < n;
       i += (long)gridDim.x * blockDim.x)
    y[i] = (_Float16)fmaxf(x[i], 0.f);
}

__global__ void cbnorm_kernel(const float* __restrict__ cb,
                              float* __restrict__ out, int K, int D) {
  int j = blockIdx.x * blockDim.x + threadIdx.x;
  if (j < K) {
    float s = 0.f;
    for (int c = 0; c < D; ++c) { float v = cb[(long)j * D + c]; s += v * v; }
    out[j] = s;
  }
}

// ---------------------------------------------------------------------------
// BatchNorm (batch statistics, matching jnp.mean/var over (N,H,W))
// ---------------------------------------------------------------------------
template <typename T>
__global__ void bn_stats_kernel(const T* __restrict__ x, int C, int HW, int N,
                                float* __restrict__ mean,
                                float* __restrict__ inv) {
  const int c = blockIdx.x;
  float s = 0.f, s2 = 0.f;
  for (int b = 0; b < N; ++b) {
    const T* p = x + ((long)b * C + c) * HW;
    for (int i = threadIdx.x; i < HW; i += blockDim.x) {
      float v = (float)p[i];
      s += v; s2 += v * v;
    }
  }
  __shared__ float sv[256], sq[256];
  sv[threadIdx.x] = s; sq[threadIdx.x] = s2;
  __syncthreads();
  for (int st = 128; st > 0; st >>= 1) {
    if (threadIdx.x < st) {
      sv[threadIdx.x] += sv[threadIdx.x + st];
      sq[threadIdx.x] += sq[threadIdx.x + st];
    }
    __syncthreads();
  }
  if (threadIdx.x == 0) {
    float cnt = (float)N * (float)HW;
    float mu = sv[0] / cnt;
    float var = sq[0] / cnt - mu * mu;
    mean[c] = mu;
    inv[c] = rsqrtf(fmaxf(var, 0.f) + 1e-5f);
  }
}

// channel index = (i >> hwShift) & cMask  (HW, C are powers of two here)
__global__ void bn_relu_f16_kernel(_Float16* x, const float* mean,
                                   const float* inv, const float* g,
                                   const float* b, long n, int hwShift,
                                   int cMask) {
  for (long i = blockIdx.x * (long)blockDim.x + threadIdx.x; i < n;
       i += (long)gridDim.x * blockDim.x) {
    int c = (int)((i >> hwShift) & cMask);
    float v = ((float)x[i] - mean[c]) * inv[c] * g[c] + b[c];
    x[i] = (_Float16)fmaxf(v, 0.f);
  }
}

__global__ void bn_relu_f32_to_f16_kernel(const float* __restrict__ x,
                                          _Float16* __restrict__ y,
                                          const float* mean, const float* inv,
                                          const float* g, const float* b,
                                          long n, int hwShift, int cMask) {
  for (long i = blockIdx.x * (long)blockDim.x + threadIdx.x; i < n;
       i += (long)gridDim.x * blockDim.x) {
    int c = (int)((i >> hwShift) & cMask);
    float v = (x[i] - mean[c]) * inv[c] * g[c] + b[c];
    y[i] = (_Float16)fmaxf(v, 0.f);
  }
}

__global__ void bn_add_f32_kernel(const float* __restrict__ x,
                                  float* __restrict__ acc, const float* mean,
                                  const float* inv, const float* g,
                                  const float* b, long n, int hwShift,
                                  int cMask) {
  for (long i = blockIdx.x * (long)blockDim.x + threadIdx.x; i < n;
       i += (long)gridDim.x * blockDim.x) {
    int c = (int)((i >> hwShift) & cMask);
    acc[i] += (x[i] - mean[c]) * inv[c] * g[c] + b[c];
  }
}

// ---------------------------------------------------------------------------
// Implicit-GEMM convolution, geometry as template params so all the im2col
// address math strength-reduces (no runtime idiv chains).
// out[m][n], m=(b,oh,ow), n=cout, K = CIN*KH*KW, W stored [Cout][K] (OIHW).
// Block: 256 thr = 8 waves -> 64x64 tile; wave owns 16x32 (2 accumulators,
// shared A fragment).  K staged 64 per barrier round -> 4 WMMA / wave / round.
// Per thread the in-tile k is FIXED (tid&63), so the (ci,kh,kw) decode is
// hoisted out of the row loop.  B staged with 16B vector copies.
// ---------------------------------------------------------------------------
template <int CIN, int HIN, int WIN, int KH, int KW, int STRIDE, int PAD,
          int HOUT, int WOUT, int COUT>
__global__ void conv_gemm_kernel(const _Float16* __restrict__ in,
                                 const _Float16* __restrict__ W,
                                 const float* __restrict__ bias, float* out32,
                                 _Float16* out16) {
  constexpr int KTOT = CIN * KH * KW;   // multiple of 8 for all our shapes
  constexpr int HWO = HOUT * WOUT;
  constexpr int KHW = KH * KW;
  __shared__ __attribute__((aligned(16))) _Float16 As[64 * 64];
  __shared__ __attribute__((aligned(16))) _Float16 Bs[64 * 64];
  const int tid = threadIdx.x;
  const int wave = tid >> 5;
  const int mt = wave & 3, nt = wave >> 2;      // 4 x 2 wave grid
  const int m0 = blockIdx.x * 64;
  const int n0 = blockIdx.y * 64;
  const int kkA = tid & 63;                     // fixed in-tile k per thread
  const int rbA = tid >> 6;                     // row phase 0..3
  v8f acc0 = {0.f, 0.f, 0.f, 0.f, 0.f, 0.f, 0.f, 0.f};
  v8f acc1 = {0.f, 0.f, 0.f, 0.f, 0.f, 0.f, 0.f, 0.f};
  for (int k0 = 0; k0 < KTOT; k0 += 64) {
    // ---- stage A: im2col gather; k-decode hoisted (loop-invariant) ----
    const int k = k0 + kkA;
    const bool kok = (k < KTOT);
    const int ci = kok ? (k / KHW) : 0;
    const int kr = k - ci * KHW;
    const int kh = kr / KW;
    const int kw = kr - kh * KW;
#pragma unroll
    for (int i = 0; i < 16; ++i) {
      const int r = i * 4 + rbA;
      const int m = m0 + r;
      const int b = m / HWO, rm = m - b * HWO;
      const int oh = rm / WOUT, ow = rm - oh * WOUT;
      const int ih = oh * STRIDE - PAD + kh;
      const int iw = ow * STRIDE - PAD + kw;
      _Float16 v = (_Float16)0.f;
      if (kok && ih >= 0 && ih < HIN && iw >= 0 && iw < WIN)
        v = in[(((long)b * CIN + ci) * HIN + ih) * WIN + iw];
      As[r * 64 + kkA] = v;
    }
    // ---- stage B: weights, 16B vector copies (KTOT % 8 == 0) ----
#pragma unroll
    for (int i = 0; i < 2; ++i) {
      int ch = i * 256 + tid;
      int r = ch >> 3, ko = (ch & 7) * 8;
      int kb = k0 + ko;
      v8h vv = zero8h();
      if (kb < KTOT) vv = *(const v8h*)&W[(long)(n0 + r) * KTOT + kb];
      *(v8h*)&Bs[r * 64 + ko] = vv;
    }
    if (k0 + 64 < KTOT)               // prefetch next weight tile rows
      __builtin_prefetch(W + (long)(n0 + (tid & 63)) * KTOT + k0 + 64, 0, 1);
    __syncthreads();
#pragma unroll
    for (int ks = 0; ks < 2; ++ks) {
      v16h af = frag_a(&As[mt * 16 * 64 + ks * 32], 64);
      v16h b0 = frag_b(&Bs[(nt * 32) * 64 + ks * 32], 64);
      v16h b1 = frag_b(&Bs[(nt * 32 + 16) * 64 + ks * 32], 64);
      acc0 = wmma16(af, b0, acc0);
      acc1 = wmma16(af, b1, acc1);
    }
    __syncthreads();
  }
  const int lane = tid & 31;
  const int c0 = n0 + nt * 32 + (lane & 15);
  const int c1 = c0 + 16;
  const float bv0 = bias ? bias[c0] : 0.f;
  const float bv1 = bias ? bias[c1] : 0.f;
#pragma unroll
  for (int j = 0; j < 8; ++j) {
    int m = m0 + mt * 16 + j + 8 * (lane >> 4);
    int b = m / HWO, rm = m - b * HWO;
    int oh = rm / WOUT, ow = rm - oh * WOUT;
    long o0 = (((long)b * COUT + c0) * HOUT + oh) * WOUT + ow;
    long o1 = (((long)b * COUT + c1) * HOUT + oh) * WOUT + ow;
    float v0 = acc0[j] + bv0, v1 = acc1[j] + bv1;
    if (out32) { out32[o0] = v0; out32[o1] = v1; }
    if (out16) { out16[o0] = (_Float16)v0; out16[o1] = (_Float16)v1; }
  }
}

// ---------------------------------------------------------------------------
// Generic WMMA GEMM: out[m][n] = sum_k A(m,k) * B[n][k] (+bias[n]).
// a_mode 0: A row-major (lda). a_mode 1: xs gather m=(t,b) from flat_st.
// a_mode 2: encoded-samples gather (f = blockIdx.z, offset p+1 read on device).
// A rows are k-contiguous in every mode -> 16B vector staging for A and B.
// Block tile 64x64 (wave = 16x32), K staged 64.  K, N multiples of 64.
// ---------------------------------------------------------------------------
__global__ void gemm_kernel(const _Float16* __restrict__ A, int a_mode, int lda,
                            const _Float16* __restrict__ flat,
                            const int* __restrict__ p_ptr,
                            const _Float16* __restrict__ B, int ldb,
                            long b_zstride, const float* __restrict__ bias,
                            float* out32, _Float16* out16, int ldc, int K) {
  __shared__ __attribute__((aligned(16))) _Float16 As[64 * 64];
  __shared__ __attribute__((aligned(16))) _Float16 Bs[64 * 64];
  const int tid = threadIdx.x;
  const int wave = tid >> 5;
  const int mt = wave & 3, nt = wave >> 2;
  const int m0 = blockIdx.x * 64;
  const int n0 = blockIdx.y * 64;
  const int z = blockIdx.z;
  const _Float16* Bz = B + (long)z * b_zstride;
  const int pval = (a_mode == 2) ? p_ptr[0] : 0;
  const _Float16* Asrc = (a_mode == 0) ? A : flat;
  v8f acc0 = {0.f, 0.f, 0.f, 0.f, 0.f, 0.f, 0.f, 0.f};
  v8f acc1 = {0.f, 0.f, 0.f, 0.f, 0.f, 0.f, 0.f, 0.f};
  for (int k0 = 0; k0 < K; k0 += 64) {
#pragma unroll
    for (int i = 0; i < 2; ++i) {      // stage A: 512 x 8-half chunks
      int ch = i * 256 + tid;
      int r = ch >> 3, ko = (ch & 7) * 8;
      int m = m0 + r;
      long src;
      if (a_mode == 0) {
        src = (long)m * lda + k0 + ko;
      } else if (a_mode == 1) {        // xs[t][b][k] = flat[b][t][k]
        int t = m >> 6, b = m & 63;
        src = ((long)b * 1024 + t) * 256 + k0 + ko;
      } else {                         // enc[f][b][k] = flat[b][p+1+f][k]
        src = ((long)m * 1024 + pval + 1 + z) * 256 + k0 + ko;
      }
      *(v8h*)&As[r * 64 + ko] = *(const v8h*)&Asrc[src];
    }
#pragma unroll
    for (int i = 0; i < 2; ++i) {      // stage B
      int ch = i * 256 + tid;
      int r = ch >> 3, ko = (ch & 7) * 8;
      *(v8h*)&Bs[r * 64 + ko] = *(const v8h*)&Bz[(long)(n0 + r) * ldb + k0 + ko];
    }
    __syncthreads();
#pragma unroll
    for (int ks = 0; ks < 2; ++ks) {
      v16h af = frag_a(&As[mt * 16 * 64 + ks * 32], 64);
      v16h b0 = frag_b(&Bs[(nt * 32) * 64 + ks * 32], 64);
      v16h b1 = frag_b(&Bs[(nt * 32 + 16) * 64 + ks * 32], 64);
      acc0 = wmma16(af, b0, acc0);
      acc1 = wmma16(af, b1, acc1);
    }
    __syncthreads();
  }
  const int lane = tid & 31;
  const int c0 = n0 + nt * 32 + (lane & 15);
  const int c1 = c0 + 16;
  const float bv0 = bias ? bias[c0] : 0.f;
  const float bv1 = bias ? bias[c1] : 0.f;
#pragma unroll
  for (int j = 0; j < 8; ++j) {
    int m = m0 + mt * 16 + j + 8 * (lane >> 4);
    long ob = (long)z * 64 * ldc + (long)m * ldc;
    float v0 = acc0[j] + bv0, v1 = acc1[j] + bv1;
    if (out32) { out32[ob + c0] = v0; out32[ob + c1] = v1; }
    if (out16) { out16[ob + c0] = (_Float16)v0; out16[ob + c1] = (_Float16)v1; }
  }
}

// ---------------------------------------------------------------------------
// Vector quantization: per 16-row stripe, WMMA-GEMM z(16x256) x cb^T(256x512),
// fused argmin of (||cb_j||^2 - 2 z.cb_j); writes z_q (NCHW f32, d_out) and
// flat_st (f16, [b][t][c]).  z read from z_e (NCHW f32 in d_out).
// ---------------------------------------------------------------------------
__global__ void vq_kernel(const float* __restrict__ ze,
                          const float* __restrict__ cb,
                          const _Float16* __restrict__ cbh,
                          const float* __restrict__ cbnorm,
                          float* __restrict__ zq, _Float16* __restrict__ flat) {
  __shared__ __attribute__((aligned(16))) _Float16 Ah[16 * 256];
  __shared__ float redv[8][32][8];
  __shared__ int   redi[8][32][8];
  __shared__ int   best[16];
  const int tid = threadIdx.x;
  const int m0 = blockIdx.x * 16;
#pragma unroll
  for (int i = 0; i < 16; ++i) {   // gather NCHW -> (m, c) f16
    int e = i * 256 + tid;
    int r = e >> 8, c = e & 255;
    int m = m0 + r;
    int b = m >> 10, rem = m & 1023, h = rem >> 5, w = rem & 31;
    Ah[r * 256 + c] = (_Float16)ze[(((long)b * 256 + c) * 32 + h) * 32 + w];
  }
  __syncthreads();
  const int wave = tid >> 5, lane = tid & 31;
  v16h af[8];
#pragma unroll
  for (int ks = 0; ks < 8; ++ks) af[ks] = frag_a(&Ah[ks * 32], 256);
  float mv[8]; int mi[8];
#pragma unroll
  for (int j = 0; j < 8; ++j) { mv[j] = 3.4e38f; mi[j] = 0; }
  for (int q = 0; q < 4; ++q) {
    const int ntile = wave + q * 8;            // covers N = 512
    v8f acc = {0.f, 0.f, 0.f, 0.f, 0.f, 0.f, 0.f, 0.f};
#pragma unroll
    for (int ks = 0; ks < 8; ++ks) {
      v16h bf = frag_b(cbh + (long)ntile * 16 * 256 + ks * 32, 256);
      acc = wmma16(af[ks], bf, acc);
    }
    const int n = ntile * 16 + (lane & 15);
    const float cn = cbnorm[n];
#pragma unroll
    for (int j = 0; j < 8; ++j) {
      float val = cn - 2.0f * acc[j];
      if (val < mv[j]) { mv[j] = val; mi[j] = n; }
    }
  }
#pragma unroll
  for (int j = 0; j < 8; ++j) { redv[wave][lane][j] = mv[j]; redi[wave][lane][j] = mi[j]; }
  __syncthreads();
  if (tid < 16) {                    // row r: candidates in lanes of half r>>3
    const int hf = tid >> 3, jr = tid & 7;
    float bv = 3.4e38f; int bi = 0;
    for (int wv = 0; wv < 8; ++wv)
      for (int l = hf * 16; l < hf * 16 + 16; ++l) {
        float v = redv[wv][l][jr];
        if (v < bv) { bv = v; bi = redi[wv][l][jr]; }
      }
    best[tid] = bi;
  }
  __syncthreads();
#pragma unroll
  for (int i = 0; i < 16; ++i) {
    int e = i * 256 + tid;
    int r = e >> 8, c = e & 255;
    int m = m0 + r;
    int idx = best[r];
    float v = cb[(long)idx * 256 + c];
    int b = m >> 10, rem = m & 1023, h = rem >> 5, w = rem & 31;
    zq[(((long)b * 256 + c) * 32 + h) * 32 + w] = v;
    flat[((long)b * 1024 + rem) * 256 + c] = (_Float16)v;
  }
}

// ---------------------------------------------------------------------------
// GRU scan: single persistent workgroup; h (64x128) lives in LDS (f32 state +
// f16 mirror for WMMA A-fragments).  Wave w owns gate columns [16w,16w+16):
// r/z/n accumulators stay in registers, so the gate math is local per lane.
// hp[b][g] = sum_k h[b][k]*w_hh[g][k] + b_hh[g];   xp includes b_ih already.
// ---------------------------------------------------------------------------
__global__ void gru_kernel(const float* __restrict__ h0,
                           const float* __restrict__ xp,
                           const _Float16* __restrict__ whh,
                           const float* __restrict__ bhh,
                           const int* __restrict__ p_ptr,
                           float* __restrict__ ct32,
                           _Float16* __restrict__ ct16) {
  __shared__ float    h32[64 * 128];
  __shared__ __attribute__((aligned(16))) _Float16 h16[64 * 128];
  const int tid = threadIdx.x;
  for (int i = tid; i < 64 * 128; i += 256) {
    float v = h0[i];
    h32[i] = v; h16[i] = (_Float16)v;
  }
  __syncthreads();
  const int wave = tid >> 5, lane = tid & 31;
  const int jcol = wave * 16 + (lane & 15);
  const int P = p_ptr[0];
  for (int t = 0; t <= P; ++t) {
    float hnew[4][8];
    for (int mtile = 0; mtile < 4; ++mtile) {
      v8f ar = {0.f,0.f,0.f,0.f,0.f,0.f,0.f,0.f};
      v8f az = {0.f,0.f,0.f,0.f,0.f,0.f,0.f,0.f};
      v8f an = {0.f,0.f,0.f,0.f,0.f,0.f,0.f,0.f};
#pragma unroll
      for (int ks = 0; ks < 4; ++ks) {
        v16h afr = frag_a(&h16[mtile * 16 * 128 + ks * 32], 128);
        v16h br = frag_b(whh + (long)(0   + wave * 16) * 128 + ks * 32, 128);
        v16h bz = frag_b(whh + (long)(128 + wave * 16) * 128 + ks * 32, 128);
        v16h bn = frag_b(whh + (long)(256 + wave * 16) * 128 + ks * 32, 128);
        ar = wmma16(afr, br, ar);
        az = wmma16(afr, bz, az);
        an = wmma16(afr, bn, an);
      }
#pragma unroll
      for (int j = 0; j < 8; ++j) {
        int m = mtile * 16 + j + 8 * (lane >> 4);
        long xb = ((long)t * 64 + m) * 384;
        float xr = xp[xb + jcol];
        float xz = xp[xb + 128 + jcol];
        float xn = xp[xb + 256 + jcol];
        float hr = ar[j] + bhh[jcol];
        float hz = az[j] + bhh[128 + jcol];
        float hn = an[j] + bhh[256 + jcol];
        float r  = 1.f / (1.f + __expf(-(xr + hr)));
        float zz = 1.f / (1.f + __expf(-(xz + hz)));
        float nn = tanhf(xn + r * hn);
        float ho = h32[m * 128 + jcol];
        hnew[mtile][j] = (1.f - zz) * nn + zz * ho;
      }
    }
    __syncthreads();   // all waves finished reading h16/h32 for this step
#pragma unroll
    for (int mtile = 0; mtile < 4; ++mtile)
#pragma unroll
      for (int j = 0; j < 8; ++j) {
        int m = mtile * 16 + j + 8 * (lane >> 4);
        h32[m * 128 + jcol] = hnew[mtile][j];
        h16[m * 128 + jcol] = (_Float16)hnew[mtile][j];
      }
    __syncthreads();   // writes visible before next step's fragment loads
  }
  for (int i = tid; i < 64 * 128; i += 256) {
    ct32[i] = h32[i];
    ct16[i] = (_Float16)h32[i];
  }
}

// ---------------------------------------------------------------------------
// Final NCE + accuracy from totals[16][64][64].
// nce = -sum_{f,b}(t[f][b][b] - lse_d t[f][b][:]) / (B*F)
// acc = (1/B) * #{ d : argmax_b (t[15][b][d] - lse15[b]) == d }
// ---------------------------------------------------------------------------
__global__ void finalize_kernel(const float* __restrict__ totals,
                                float* __restrict__ out) {
  __shared__ float partial[256];
  __shared__ float lse15[64];
  __shared__ int corr;
  const int tid = threadIdx.x;
  float local = 0.f;
  for (int row = tid; row < 1024; row += 256) {
    int f = row >> 6, b = row & 63;
    const float* t = totals + ((long)f * 64 + b) * 64;
    float mx = t[0];
    for (int d = 1; d < 64; ++d) mx = fmaxf(mx, t[d]);
    float s = 0.f;
    for (int d = 0; d < 64; ++d) s += __expf(t[d] - mx);
    float lse = mx + __logf(s);
    local += t[b] - lse;
    if (f == 15) lse15[b] = lse;
  }
  partial[tid] = local;
  if (tid == 0) corr = 0;
  __syncthreads();
  for (int st = 128; st > 0; st >>= 1) {
    if (tid < st) partial[tid] += partial[tid + st];
    __syncthreads();
  }
  if (tid == 0) out[1] = -partial[0] / 1024.0f;
  if (tid < 64) {
    int d = tid;
    float bv = -3.4e38f; int bb = 0;
    for (int b = 0; b < 64; ++b) {
      float v = totals[((long)15 * 64 + b) * 64 + d] - lse15[b];
      if (v > bv) { bv = v; bb = b; }
    }
    if (bb == d) atomicAdd(&corr, 1);
  }
  __syncthreads();
  if (tid == 0) out[0] = (float)corr / 64.0f;
}

// ---------------------------------------------------------------------------
// Host orchestration
// ---------------------------------------------------------------------------
extern "C" void kernel_launch(void* const* d_in, const int* in_sizes, int n_in,
                              void* d_out, int out_size, void* d_ws,
                              size_t ws_size, hipStream_t stream) {
  (void)in_sizes; (void)n_in; (void)out_size; (void)ws_size;
  const float* x         = (const float*)d_in[0];
  const float* hidden    = (const float*)d_in[1];
  const float* conv1_w   = (const float*)d_in[2];
  const float* conv1_b   = (const float*)d_in[3];
  const float* bn1_g     = (const float*)d_in[4];
  const float* bn1_b     = (const float*)d_in[5];
  const float* conv2_w   = (const float*)d_in[6];
  const float* conv2_b   = (const float*)d_in[7];
  const float* res_c3_w  = (const float*)d_in[8];
  const float* res_c3_b  = (const float*)d_in[9];
  const float* res_bn1_g = (const float*)d_in[10];
  const float* res_bn1_b = (const float*)d_in[11];
  const float* res_c1_w  = (const float*)d_in[12];
  const float* res_c1_b  = (const float*)d_in[13];
  const float* res_bn2_g = (const float*)d_in[14];
  const float* res_bn2_b = (const float*)d_in[15];
  const float* codebook  = (const float*)d_in[16];
  const float* gru_w_ih  = (const float*)d_in[17];
  const float* gru_w_hh  = (const float*)d_in[18];
  const float* gru_b_ih  = (const float*)d_in[19];
  const float* gru_b_hh  = (const float*)d_in[20];
  const float* wk_w      = (const float*)d_in[21];
  const float* wk_b      = (const float*)d_in[22];
  const int*   p_sample  = (const int*)d_in[23];

  float* out = (float*)d_out;
  float* ze  = out + 2;                 // z_e_x  (64,256,32,32) — built in place
  float* zq  = out + 2 + 16777216L;     // z_q_x_bar

  char* base = (char*)d_ws;
  size_t off = 0;
  auto alloc = [&](size_t bytes) -> void* {
    void* p = base + off;
    off += (bytes + 255) & ~(size_t)255;
    return p;
  };
  _Float16* xh    = (_Float16*)alloc(3145728L * 2);        // x f16
  _Float16* w1h   = (_Float16*)alloc(12288L * 2);
  _Float16* w2h   = (_Float16*)alloc(1048576L * 2);
  _Float16* w3h   = (_Float16*)alloc(1179648L * 2);        // (2,256,2304)
  _Float16* wr1h  = (_Float16*)alloc(131072L * 2);         // (2,256,256)
  _Float16* cbh   = (_Float16*)alloc(131072L * 2);
  _Float16* wihh  = (_Float16*)alloc(98304L * 2);
  _Float16* whhh  = (_Float16*)alloc(49152L * 2);
  _Float16* wkwh  = (_Float16*)alloc(524288L * 2);         // (4096,128)
  _Float16* h1h   = (_Float16*)alloc(67108864L * 2);       // conv1 act (64,256,64,64)
  _Float16* bufA  = (_Float16*)alloc(16777216L * 2);       // f16 act (32x32)
  float*    bufT  = (float*)   alloc(16777216L * 4);       // f32 tmp (32x32)
  _Float16* flath = (_Float16*)alloc(16777216L * 2);       // flat_st (64,1024,256)
  float*    xp    = (float*)   alloc(25165824L * 4);       // (1024,64,384)
  float*    ct32  = (float*)   alloc(8192L * 4);
  _Float16* ct16  = (_Float16*)alloc(8192L * 2);
  _Float16* predh = (_Float16*)alloc(262144L * 2);         // (64,4096)=[b][f*256+c]
  float*    tot   = (float*)   alloc(65536L * 4);          // (16,64,64)
  float*    meanb = (float*)   alloc(256L * 4);
  float*    invb  = (float*)   alloc(256L * 4);
  float*    cbn   = (float*)   alloc(512L * 4);

  const dim3 blk(256);
  const dim3 gcv(4096);

  // fp16 conversions of inputs / weights
  cvt_f32_f16_kernel<<<gcv, blk, 0, stream>>>(x, xh, 3145728L);
  cvt_f32_f16_kernel<<<64, blk, 0, stream>>>(conv1_w, w1h, 12288L);
  cvt_f32_f16_kernel<<<gcv, blk, 0, stream>>>(conv2_w, w2h, 1048576L);
  cvt_f32_f16_kernel<<<gcv, blk, 0, stream>>>(res_c3_w, w3h, 1179648L);
  cvt_f32_f16_kernel<<<512, blk, 0, stream>>>(res_c1_w, wr1h, 131072L);
  cvt_f32_f16_kernel<<<512, blk, 0, stream>>>(codebook, cbh, 131072L);
  cvt_f32_f16_kernel<<<384, blk, 0, stream>>>(gru_w_ih, wihh, 98304L);
  cvt_f32_f16_kernel<<<192, blk, 0, stream>>>(gru_w_hh, whhh, 49152L);
  cvt_f32_f16_kernel<<<2048, blk, 0, stream>>>(wk_w, wkwh, 524288L);
  cbnorm_kernel<<<2, blk, 0, stream>>>(codebook, cbn, 512, 256);

  // conv1: (64,3,128,128) -> (64,256,64,64), k4 s2 p1 ; K=48
  conv_gemm_kernel<3, 128, 128, 4, 4, 2, 1, 64, 64, 256>
      <<<dim3(4096, 4), blk, 0, stream>>>(xh, w1h, conv1_b, nullptr, h1h);
  bn_stats_kernel<_Float16><<<256, blk, 0, stream>>>(h1h, 256, 4096, 64, meanb, invb);
  bn_relu_f16_kernel<<<gcv, blk, 0, stream>>>(h1h, meanb, invb, bn1_g, bn1_b,
                                              67108864L, 12, 255);

  // conv2: (64,256,64,64) -> z_e (64,256,32,32), k4 s2 p1 ; K=4096
  conv_gemm_kernel<256, 64, 64, 4, 4, 2, 1, 32, 32, 256>
      <<<dim3(1024, 4), blk, 0, stream>>>(h1h, w2h, conv2_b, ze, nullptr);

  // 2x residual blocks
  for (int r = 0; r < 2; ++r) {
    relu_f32_to_f16_kernel<<<gcv, blk, 0, stream>>>(ze, bufA, 16777216L);
    conv_gemm_kernel<256, 32, 32, 3, 3, 1, 1, 32, 32, 256>
        <<<dim3(1024, 4), blk, 0, stream>>>(bufA, w3h + (long)r * 589824,
                                            res_c3_b + r * 256, bufT, nullptr);
    bn_stats_kernel<float><<<256, blk, 0, stream>>>(bufT, 256, 1024, 64, meanb, invb);
    bn_relu_f32_to_f16_kernel<<<gcv, blk, 0, stream>>>(
        bufT, bufA, meanb, invb, res_bn1_g + r * 256, res_bn1_b + r * 256,
        16777216L, 10, 255);
    conv_gemm_kernel<256, 32, 32, 1, 1, 1, 0, 32, 32, 256>
        <<<dim3(1024, 4), blk, 0, stream>>>(bufA, wr1h + (long)r * 65536,
                                            res_c1_b + r * 256, bufT, nullptr);
    bn_stats_kernel<float><<<256, blk, 0, stream>>>(bufT, 256, 1024, 64, meanb, invb);
    bn_add_f32_kernel<<<gcv, blk, 0, stream>>>(
        bufT, ze, meanb, invb, res_bn2_g + r * 256, res_bn2_b + r * 256,
        16777216L, 10, 255);
  }

  // Vector quantization (writes z_q to d_out and flat_st f16)
  vq_kernel<<<4096, blk, 0, stream>>>(ze, codebook, cbh, cbn, zq, flath);

  // xp = xs @ w_ih^T + b_ih, computed for all 1024 timesteps (static grid)
  gemm_kernel<<<dim3(1024, 6, 1), blk, 0, stream>>>(
      nullptr, 1, 0, flath, p_sample, wihh, 256, 0, gru_b_ih, xp, nullptr, 384, 256);

  // GRU scan (t = 0..p_sample), persistent single workgroup
  gru_kernel<<<1, blk, 0, stream>>>(hidden, xp, whhh, gru_b_hh, p_sample, ct32, ct16);

  // pred[b][f*256+c] = ct[b] . wk_w[f*256+c] + wk_b   (f16 for totals GEMM)
  gemm_kernel<<<dim3(1, 64, 1), blk, 0, stream>>>(
      ct16, 0, 128, nullptr, p_sample, wkwh, 128, 0, wk_b, nullptr, predh, 4096, 128);

  // totals[f][b][d] = enc[f][b][:] . pred[f][d][:]
  gemm_kernel<<<dim3(1, 1, 16), blk, 0, stream>>>(
      nullptr, 2, 0, flath, p_sample, predh, 4096, 256, nullptr, tot, nullptr, 64, 256);

  // accuracy + nce
  finalize_kernel<<<1, blk, 0, stream>>>(tot, out);
}
